// MultiHeadedDotAttention_8108898255381
// MI455X (gfx1250) — matrix operand order repaired and checked by reading, not compiled, not run on hardware
//
#include <hip/hip_runtime.h>

// ---------------------------------------------------------------------------
// MultiHeadedDotAttention + AoA for MI455X (gfx1250), bf16 WMMA pipeline.
// B=8, S=1024, D=1024, H=16, DK=64.
// Register-blocked wave-tile GEMMs: 2x4 (projections), 2x2 dual (AoA),
// 2-q-tile flash attention.
// ---------------------------------------------------------------------------

typedef __attribute__((ext_vector_type(16))) __bf16 v16bf;
typedef __attribute__((ext_vector_type(8)))  __bf16 v8bf;
typedef __attribute__((ext_vector_type(8)))  float  v8f;

#define NEGV (-1.0e9f)

static __device__ inline v8f wmma_bf16(v16bf a, v16bf b, v8f c) {
  // D = A(16x32 bf16) * B(32x16 bf16) + C(16x16 f32)
  return __builtin_amdgcn_wmma_f32_16x16x32_bf16(
      /*neg_a=*/false, a, /*neg_b=*/false, b,
      /*c_mod=*/(short)0, c, /*reuse_a=*/false, /*reuse_b=*/false);
}

// A-matrix (16x32, row-major source, ld in elements).
// Lane l: m = l&15, kh = l>>4; holds K = kh*8..+7 and 16+kh*8..+7.
static __device__ inline v16bf load_A16(const __bf16* base, int ld) {
  const int l  = threadIdx.x & 31;
  const int m  = l & 15;
  const int kh = l >> 4;
  const __bf16* p = base + (size_t)m * ld + kh * 8;
  v8bf lo = *(const v8bf*)(p);
  v8bf hi = *(const v8bf*)(p + 16);
  return __builtin_shufflevector(lo, hi, 0,1,2,3,4,5,6,7,8,9,10,11,12,13,14,15);
}

// B-matrix (32x16): column n comes from source row n.
// Lane l: n = l&15, kh = l>>4; holds K = kh*8..+7 and 16+kh*8..+7.
static __device__ inline v16bf load_B16(const __bf16* base, int ld) {
  const int l  = threadIdx.x & 31;
  const int n  = l & 15;
  const int kh = l >> 4;
  const __bf16* p = base + (size_t)n * ld + kh * 8;
  v8bf lo = *(const v8bf*)(p);
  v8bf hi = *(const v8bf*)(p + 16);
  return __builtin_shufflevector(lo, hi, 0,1,2,3,4,5,6,7,8,9,10,11,12,13,14,15);
}

// ---------------------------------------------------------------------------
// f32 -> bf16 conversion
// ---------------------------------------------------------------------------
__global__ void cvt_f32_bf16(const float* __restrict__ x, __bf16* __restrict__ y, int n) {
  int i = blockIdx.x * 256 + threadIdx.x;
  if (i < n) y[i] = (__bf16)x[i];
}

// ---------------------------------------------------------------------------
// Y = X * W^T + bias.  X:[8192,1024] bf16, W:[1024,1024] bf16 (row=out feat).
// Wave computes a 32x64 block: 2 M-tiles x 4 N-tiles, K=1024.
// mode 0: out[((b*16+h)*1024 + s)*64 + dk]  ([B,H,S,DK] for Q,K)
// mode 1: out[((b*16+h)*64 + dk)*1024 + s]  ([B,H,DK,S] transposed, for V)
// ---------------------------------------------------------------------------
__global__ void qkv_gemm(const __bf16* __restrict__ X, const __bf16* __restrict__ W,
                         const float* __restrict__ bias, __bf16* __restrict__ out,
                         int mode) {
  const int m0 = blockIdx.x * 32;
  const int n0 = blockIdx.y * 64;
  const int l  = threadIdx.x & 31;
  const int hv = l >> 4;
  const int n  = l & 15;

  v8f acc[2][4];
#pragma unroll
  for (int i = 0; i < 2; ++i)
#pragma unroll
    for (int j = 0; j < 4; ++j) acc[i][j] = (v8f){};

  const __bf16* a0 = X + (size_t)m0 * 1024;
  const __bf16* a1 = X + (size_t)(m0 + 16) * 1024;
  for (int kk = 0; kk < 1024; kk += 32) {
    __builtin_prefetch(a0 + kk + 512, 0, 0);   // global_prefetch_b8
    v16bf A0 = load_A16(a0 + kk, 1024);
    v16bf A1 = load_A16(a1 + kk, 1024);
#pragma unroll
    for (int j = 0; j < 4; ++j) {
      v16bf Bj = load_B16(W + (size_t)(n0 + 16 * j) * 1024 + kk, 1024);
      acc[0][j] = wmma_bf16(A0, Bj, acc[0][j]);
      acc[1][j] = wmma_bf16(A1, Bj, acc[1][j]);
    }
  }

#pragma unroll
  for (int j = 0; j < 4; ++j) {
    const int col = n0 + 16 * j + n;
    const float bs = bias[col];
    const int h  = col >> 6;
    const int dk = col & 63;
#pragma unroll
    for (int i = 0; i < 2; ++i) {
#pragma unroll
      for (int r = 0; r < 8; ++r) {
        const int m  = m0 + 16 * i + r + 8 * hv;
        const int bb = m >> 10;       // batch
        const int s  = m & 1023;      // sequence pos
        const float v = acc[i][j][r] + bs;
        if (mode == 0)
          out[(((size_t)(bb * 16 + h)) * 1024 + s) * 64 + dk] = (__bf16)v;
        else
          out[(((size_t)(bb * 16 + h)) * 64 + dk) * 1024 + s] = (__bf16)v;
      }
    }
  }
}

// ---------------------------------------------------------------------------
// Flash-style attention: one wave per (b, h, 32-row q block = 2 tiles).
// K/V panel loads are shared across both q-tiles.
// ---------------------------------------------------------------------------
__global__ void attention(const __bf16* __restrict__ Qh, const __bf16* __restrict__ Kh,
                          const __bf16* __restrict__ Vt, const int* __restrict__ amask,
                          __bf16* __restrict__ att) {
  __shared__ __bf16 lds[2][16 * 32];   // P tile staging (C-layout -> A-layout)

  const int qt = blockIdx.x & 31;      // S/32 = 32 q blocks
  const int bh = blockIdx.x >> 5;      // b*H + h
  const int b  = bh >> 4;
  const int h  = bh & 15;
  const int q0 = qt * 32;
  const int l  = threadIdx.x & 31;
  const int hv = l >> 4;
  const int n  = l & 15;

  v16bf aq[2][2];
#pragma unroll
  for (int qi = 0; qi < 2; ++qi) {
    const __bf16* qb = Qh + ((size_t)bh * 1024 + q0 + 16 * qi) * 64;
    aq[qi][0] = load_A16(qb + 0,  64);
    aq[qi][1] = load_A16(qb + 32, 64);
  }

  float mrow[2][8], lrow[2][8];
  v8f O[2][4];
#pragma unroll
  for (int qi = 0; qi < 2; ++qi) {
#pragma unroll
    for (int i = 0; i < 8; ++i) { mrow[qi][i] = -3.0e38f; lrow[qi][i] = 0.0f; }
#pragma unroll
    for (int t = 0; t < 4; ++t) O[qi][t] = (v8f){};
  }

  const int* maskp = amask + b * 1024;

  for (int key0 = 0; key0 < 1024; key0 += 32) {
    // ---- K panel: 4 B-tiles (2 key sub-tiles x 2 K-halves of DK) ----
    v16bf bk[2][2];
#pragma unroll
    for (int st = 0; st < 2; ++st) {
      const __bf16* kb = Kh + ((size_t)bh * 1024 + key0 + 16 * st) * 64;
      bk[st][0] = load_B16(kb + 0,  64);
      bk[st][1] = load_B16(kb + 32, 64);
    }
    const bool msk0 = (maskp[key0 + n]      == 1);
    const bool msk1 = (maskp[key0 + 16 + n] == 1);

#pragma unroll
    for (int qi = 0; qi < 2; ++qi) {
      v8f c0 = {}, c1 = {};
      c0 = wmma_bf16(aq[qi][0], bk[0][0], c0);
      c0 = wmma_bf16(aq[qi][1], bk[0][1], c0);
      c1 = wmma_bf16(aq[qi][0], bk[1][0], c1);
      c1 = wmma_bf16(aq[qi][1], bk[1][1], c1);

#pragma unroll
      for (int i = 0; i < 8; ++i) {
        c0[i] = msk0 ? NEGV : c0[i] * 0.125f;   // 1/sqrt(64)
        c1[i] = msk1 ? NEGV : c1[i] * 0.125f;
      }

      // online softmax; row reductions across half-wave of 16 lanes
#pragma unroll
      for (int i = 0; i < 8; ++i) {
        float rm = fmaxf(c0[i], c1[i]);
        for (int off = 1; off < 16; off <<= 1) rm = fmaxf(rm, __shfl_xor(rm, off, 16));
        const float mnew  = fmaxf(mrow[qi][i], rm);
        const float scale = __expf(mrow[qi][i] - mnew);
        const float p0 = __expf(c0[i] - mnew);
        const float p1 = __expf(c1[i] - mnew);
        float rs = p0 + p1;
        for (int off = 1; off < 16; off <<= 1) rs += __shfl_xor(rs, off, 16);
        lrow[qi][i] = lrow[qi][i] * scale + rs;
        mrow[qi][i] = mnew;
#pragma unroll
        for (int t = 0; t < 4; ++t) O[qi][t][i] *= scale;
        lds[qi][(i + 8 * hv) * 32 + n]      = (__bf16)p0;
        lds[qi][(i + 8 * hv) * 32 + 16 + n] = (__bf16)p1;
      }
    }
    __syncthreads();

    // ---- O += P(16x32)*V(32x16); V B-tiles shared across both q-tiles ----
    const v16bf pa0 = load_A16(&lds[0][0], 32);
    const v16bf pa1 = load_A16(&lds[1][0], 32);
#pragma unroll
    for (int t = 0; t < 4; ++t) {
      v16bf bv = load_B16(Vt + ((size_t)bh * 64 + t * 16) * 1024 + key0, 1024);
      O[0][t] = wmma_bf16(pa0, bv, O[0][t]);
      O[1][t] = wmma_bf16(pa1, bv, O[1][t]);
    }
    __syncthreads();
  }

  // ---- normalize, store attended as bf16 [B, S, D] ----
#pragma unroll
  for (int qi = 0; qi < 2; ++qi) {
#pragma unroll
    for (int i = 0; i < 8; ++i) {
      const float inv = 1.0f / lrow[qi][i];
      const int s = q0 + 16 * qi + i + 8 * hv;
      const size_t rowoff = ((size_t)b * 1024 + s) * 1024 + h * 64;
#pragma unroll
      for (int t = 0; t < 4; ++t)
        att[rowoff + t * 16 + n] = (__bf16)(O[qi][t][i] * inv);
    }
  }
}

// ---------------------------------------------------------------------------
// AoA: h = [attended, query] * W_aoa^T + b_aoa (K = 2048), GLU fuse.
// Wave computes 32 rows x 32 'a' cols (+ matching 'g' cols): 2x2 dual tiles.
// ---------------------------------------------------------------------------
__global__ void aoa_glu(const __bf16* __restrict__ att, const __bf16* __restrict__ qb,
                        const __bf16* __restrict__ Waoa, const float* __restrict__ baoa,
                        __bf16* __restrict__ out) {
  const int m0 = blockIdx.x * 32;
  const int n0 = blockIdx.y * 32;
  const int l  = threadIdx.x & 31;
  const int hv = l >> 4;
  const int n  = l & 15;

  v8f cA[2][2], cG[2][2];
#pragma unroll
  for (int i = 0; i < 2; ++i)
#pragma unroll
    for (int j = 0; j < 2; ++j) { cA[i][j] = (v8f){}; cG[i][j] = (v8f){}; }

  for (int kk = 0; kk < 2048; kk += 32) {
    const __bf16* base = (kk < 1024) ? att : qb;
    const int koff = (kk < 1024) ? kk : (kk - 1024);
    const __bf16* s0 = base + (size_t)m0 * 1024 + koff;
    const __bf16* s1 = base + (size_t)(m0 + 16) * 1024 + koff;
    __builtin_prefetch(s0 + 512, 0, 0);
    v16bf A0 = load_A16(s0, 1024);
    v16bf A1 = load_A16(s1, 1024);
#pragma unroll
    for (int j = 0; j < 2; ++j) {
      v16bf B1 = load_B16(Waoa + (size_t)(n0 + 16 * j) * 2048 + kk, 2048);
      v16bf B2 = load_B16(Waoa + (size_t)(1024 + n0 + 16 * j) * 2048 + kk, 2048);
      cA[0][j] = wmma_bf16(A0, B1, cA[0][j]);
      cA[1][j] = wmma_bf16(A1, B1, cA[1][j]);
      cG[0][j] = wmma_bf16(A0, B2, cG[0][j]);
      cG[1][j] = wmma_bf16(A1, B2, cG[1][j]);
    }
  }

#pragma unroll
  for (int j = 0; j < 2; ++j) {
    const float bA = baoa[n0 + 16 * j + n];
    const float bG = baoa[1024 + n0 + 16 * j + n];
#pragma unroll
    for (int i = 0; i < 2; ++i) {
#pragma unroll
      for (int r = 0; r < 8; ++r) {
        const float aV = cA[i][j][r] + bA;
        const float gV = cG[i][j][r] + bG;
        const float sg = 1.0f / (1.0f + __expf(-gV));
        const int m = m0 + 16 * i + r + 8 * hv;
        out[(size_t)m * 1024 + n0 + 16 * j + n] = (__bf16)(aV * sg);
      }
    }
  }
}

// ---------------------------------------------------------------------------
// Output projection: out = aoa * Wo^T + bo (f32 result). 2x4 blocking.
// ---------------------------------------------------------------------------
__global__ void out_proj(const __bf16* __restrict__ aoa, const __bf16* __restrict__ Wo,
                         const float* __restrict__ bo, float* __restrict__ out) {
  const int m0 = blockIdx.x * 32;
  const int n0 = blockIdx.y * 64;
  const int l  = threadIdx.x & 31;
  const int hv = l >> 4;
  const int n  = l & 15;

  v8f acc[2][4];
#pragma unroll
  for (int i = 0; i < 2; ++i)
#pragma unroll
    for (int j = 0; j < 4; ++j) acc[i][j] = (v8f){};

  const __bf16* a0 = aoa + (size_t)m0 * 1024;
  const __bf16* a1 = aoa + (size_t)(m0 + 16) * 1024;
  for (int kk = 0; kk < 1024; kk += 32) {
    __builtin_prefetch(a0 + kk + 512, 0, 0);
    v16bf A0 = load_A16(a0 + kk, 1024);
    v16bf A1 = load_A16(a1 + kk, 1024);
#pragma unroll
    for (int j = 0; j < 4; ++j) {
      v16bf Bj = load_B16(Wo + (size_t)(n0 + 16 * j) * 1024 + kk, 1024);
      acc[0][j] = wmma_bf16(A0, Bj, acc[0][j]);
      acc[1][j] = wmma_bf16(A1, Bj, acc[1][j]);
    }
  }

#pragma unroll
  for (int j = 0; j < 4; ++j) {
    const float bs = bo[n0 + 16 * j + n];
#pragma unroll
    for (int i = 0; i < 2; ++i) {
#pragma unroll
      for (int r = 0; r < 8; ++r) {
        const int m = m0 + 16 * i + r + 8 * hv;
        out[(size_t)m * 1024 + n0 + 16 * j + n] = acc[i][j][r] + bs;
      }
    }
  }
}

// ---------------------------------------------------------------------------
extern "C" void kernel_launch(void* const* d_in, const int* in_sizes, int n_in,
                              void* d_out, int out_size, void* d_ws, size_t ws_size,
                              hipStream_t stream) {
  const float* query = (const float*)d_in[0];
  const float* keyi  = (const float*)d_in[1];
  const float* value = (const float*)d_in[2];
  const int*   amask = (const int*)d_in[3];
  const float* Wq = (const float*)d_in[4];   const float* bq = (const float*)d_in[5];
  const float* Wk = (const float*)d_in[6];   const float* bk = (const float*)d_in[7];
  const float* Wv = (const float*)d_in[8];   const float* bv = (const float*)d_in[9];
  const float* Wa = (const float*)d_in[10];  const float* ba = (const float*)d_in[11];
  const float* Wo = (const float*)d_in[12];  const float* bo = (const float*)d_in[13];
  float* out = (float*)d_out;

  char* ws = (char*)d_ws;
  const size_t MB = 1024ull * 1024ull;
  __bf16* q_bf   = (__bf16*)(ws +   0 * MB);   // [8192,1024] bf16 query
  __bf16* k_bf   = (__bf16*)(ws +  16 * MB);
  __bf16* v_bf   = (__bf16*)(ws +  32 * MB);
  __bf16* Qh     = (__bf16*)(ws +  48 * MB);   // [B,H,S,DK]
  __bf16* Kh     = (__bf16*)(ws +  64 * MB);   // [B,H,S,DK]
  __bf16* Vt     = (__bf16*)(ws +  80 * MB);   // [B,H,DK,S]
  __bf16* attb   = (__bf16*)(ws +  96 * MB);   // attended [B,S,D]
  __bf16* aoab   = (__bf16*)(ws + 112 * MB);   // GLU out  [B,S,D]
  __bf16* Wq_bf  = (__bf16*)(ws + 128 * MB);
  __bf16* Wk_bf  = (__bf16*)(ws + 130 * MB);
  __bf16* Wv_bf  = (__bf16*)(ws + 132 * MB);
  __bf16* Wo_bf  = (__bf16*)(ws + 134 * MB);
  __bf16* Wa_bf  = (__bf16*)(ws + 136 * MB);   // 8MB

  const int NACT = 8 * 1024 * 1024;   // B*S*D
  const int NW   = 1024 * 1024;       // D*D
  const int NWA  = 4 * 1024 * 1024;   // (2D)^2

  cvt_f32_bf16<<<NACT / 256, 256, 0, stream>>>(query, q_bf, NACT);
  cvt_f32_bf16<<<NACT / 256, 256, 0, stream>>>(keyi,  k_bf, NACT);
  cvt_f32_bf16<<<NACT / 256, 256, 0, stream>>>(value, v_bf, NACT);
  cvt_f32_bf16<<<NW   / 256, 256, 0, stream>>>(Wq, Wq_bf, NW);
  cvt_f32_bf16<<<NW   / 256, 256, 0, stream>>>(Wk, Wk_bf, NW);
  cvt_f32_bf16<<<NW   / 256, 256, 0, stream>>>(Wv, Wv_bf, NW);
  cvt_f32_bf16<<<NW   / 256, 256, 0, stream>>>(Wo, Wo_bf, NW);
  cvt_f32_bf16<<<NWA  / 256, 256, 0, stream>>>(Wa, Wa_bf, NWA);

  dim3 blk(32);                  // one wave32 per workgroup; EXEC all-ones
  dim3 gproj(256, 16);           // (B*S)/32  x  D/64
  qkv_gemm<<<gproj, blk, 0, stream>>>(q_bf, Wq_bf, bq, Qh, 0);
  qkv_gemm<<<gproj, blk, 0, stream>>>(k_bf, Wk_bf, bk, Kh, 0);
  qkv_gemm<<<gproj, blk, 0, stream>>>(v_bf, Wv_bf, bv, Vt, 1);

  attention<<<dim3(8 * 16 * 32), blk, 0, stream>>>(Qh, Kh, Vt, amask, attb);

  aoa_glu<<<dim3(256, 32), blk, 0, stream>>>(attb, q_bf, Wa_bf, ba, aoab);
  out_proj<<<gproj, blk, 0, stream>>>(aoab, Wo_bf, bo, out);

  (void)in_sizes; (void)n_in; (void)out_size; (void)ws_size;
}
